// inp_GAT_39264591020281
// MI455X (gfx1250) — matrix operand order repaired and checked by reading, not compiled, not run on hardware
//
#include <hip/hip_runtime.h>
#include <hip/hip_bf16.h>
#include <math.h>

typedef __attribute__((ext_vector_type(16))) __bf16 v16bf;
typedef __attribute__((ext_vector_type(8)))  float  v8f;

#define NJ    21
#define NHIN  7
#define NHEAD 4
#define HD    64
#define INC   147
#define OUTC  (NJ*HD)      // 1344
#define LALPHA 0.2f
#define NTOK  16384        // 16 * 1024

typedef __attribute__((address_space(1))) int gint;
typedef __attribute__((address_space(3))) int lint;

union FragAB { v16bf v; unsigned u[8]; };
union FragC  { v8f  v; float    f[8]; };

__device__ __forceinline__ __bf16 tobf(float x) { return (__bf16)x; }

__device__ __forceinline__ void async_wait0() {
#if __has_builtin(__builtin_amdgcn_s_wait_asynccnt)
    __builtin_amdgcn_s_wait_asynccnt(0);
#else
    asm volatile("s_wait_asynccnt 0" ::: "memory");
#endif
}

// ---- prep: W_out (256x64 f32, [n][d]) -> Wt (64x256 bf16, [d][n]) ----
__global__ void prep_wout(const float* __restrict__ Wout, __bf16* __restrict__ Wt) {
    int i = blockIdx.x * blockDim.x + threadIdx.x;
    if (i < 256 * 64) {
        int n = i >> 6, d = i & 63;
        Wt[d * 256 + n] = (__bf16)Wout[n * 64 + d];
    }
}

// ---- main: one wave32 per token, 2 waves per block ----
__global__ void __launch_bounds__(64)
gat_kernel(const float* __restrict__ inp,    // (16,1024,147)
           const float* __restrict__ Wh,     // (4,7,64)
           const float* __restrict__ ah,     // (4,128)
           const float* __restrict__ aout,   // (128)
           const __bf16* __restrict__ Wt,    // (64,256) bf16 (transposed W_out)
           float* __restrict__ out)          // (16,1024,1344)
{
    __shared__ float  raw[2][152];           // async-staged input row
    __shared__ float  xlds[2][NJ][8];        // x, padded to 8 cols
    __shared__ __bf16 ht[2][HD][32];         // h (d-major), j padded to 32
    __shared__ __bf16 attnb[2][32][32];      // attention weights, padded
    __shared__ __bf16 xcat[2][32][256];      // concatenated heads (bf16 A-matrix)
    __shared__ float  obuf[2][32][64];       // elu(hp2) before log_softmax
    __shared__ float  f1a[2][32], f2a[2][32], rowl[2][32];

    const int wave = threadIdx.x >> 5;
    const int lane = threadIdx.x & 31;
    const int t    = blockIdx.x * 2 + wave;
    const float* xin = inp + (size_t)t * INC;

    // ---- async-stage this token's input row into LDS (ASYNCcnt path) ----
#if __has_builtin(__builtin_amdgcn_global_load_async_to_lds_b32)
    for (int i = lane; i < INC; i += 32)
        __builtin_amdgcn_global_load_async_to_lds_b32(
            (gint*)(xin + i), (lint*)(&raw[wave][i]), 0, 0);
    async_wait0();
#else
    for (int i = lane; i < INC; i += 32) raw[wave][i] = xin[i];
#endif

    // zero padded LDS regions once (ht cols >=21 and attnb rows/cols >=21 must be 0)
    {
        unsigned* p = (unsigned*)&ht[wave][0][0];
        for (int i = lane; i < HD * 32 / 2; i += 32) p[i] = 0u;
        unsigned* q = (unsigned*)&attnb[wave][0][0];
        for (int i = lane; i < 32 * 32 / 2; i += 32) q[i] = 0u;
    }
    __syncthreads();

    // build x = concat(off, loc, S): 21 x 7 (padded col 7 = 0)
    for (int i = lane; i < NJ * 8; i += 32) {
        int j = i >> 3, n = i & 7;
        float v = 0.f;
        if (n < 3)       v = raw[wave][j * 3 + n];
        else if (n < 6)  v = raw[wave][63 + j * 3 + (n - 3)];
        else if (n == 6) v = raw[wave][126 + j];
        xlds[wave][j][n] = v;
    }
    __syncthreads();

    // ---------------- per-head GAT ----------------
    for (int head = 0; head < NHEAD; ++head) {
        // h = x * W_heads[head]  (21x7 * 7x64) -> ht[d][j]
        for (int idx = lane; idx < NJ * HD; idx += 32) {
            int j = idx / HD, d = idx % HD;
            float s = 0.f;
            #pragma unroll
            for (int n = 0; n < NHIN; ++n)
                s += xlds[wave][j][n] * Wh[(head * NHIN + n) * HD + d];
            ht[wave][d][j] = tobf(s);
        }
        __syncthreads();

        // f1/f2 dot products
        if (lane < NJ) {
            float s1 = 0.f, s2 = 0.f;
            for (int d = 0; d < HD; ++d) {
                float hv = (float)ht[wave][d][lane];
                s1 += hv * ah[head * 2 * HD + d];
                s2 += hv * ah[head * 2 * HD + HD + d];
            }
            f1a[wave][lane] = s1;
            f2a[wave][lane] = s2;
        }
        __syncthreads();

        // leaky-relu + row softmax -> attnb (bf16)
        if (lane < NJ) {
            float fi = f1a[wave][lane];
            float m = -1e30f;
            for (int j = 0; j < NJ; ++j) {
                float v = fi + f2a[wave][j];
                float lr = v > 0.f ? v : LALPHA * v;
                m = fmaxf(m, lr);
            }
            float s = 0.f;
            for (int j = 0; j < NJ; ++j) {
                float v = fi + f2a[wave][j];
                float lr = v > 0.f ? v : LALPHA * v;
                s += __expf(lr - m);
            }
            float inv = 1.f / s;
            for (int j = 0; j < NJ; ++j) {
                float v = fi + f2a[wave][j];
                float lr = v > 0.f ? v : LALPHA * v;
                attnb[wave][lane][j] = tobf(__expf(lr - m) * inv);
            }
        }
        __syncthreads();

        // attend: (32x32 attn) x (32x64 h) via WMMA bf16, elu -> xcat[:, head*64:...]
        FragAB afrag[2];
        #pragma unroll
        for (int mt = 0; mt < 2; ++mt) {
            int row = mt * 16 + (lane & 15);
            #pragma unroll
            for (int v = 0; v < 8; ++v) {
                int k = 2 * v + (v >= 4 ? 8 : 0) + ((lane >> 4) << 3);
                afrag[mt].u[v] = *(const unsigned*)&attnb[wave][row][k];
            }
        }
        #pragma unroll
        for (int nt = 0; nt < 4; ++nt) {
            FragAB bfrag;
            int col = nt * 16 + (lane & 15);
            #pragma unroll
            for (int v = 0; v < 8; ++v) {
                int k = 2 * v + (v >= 4 ? 8 : 0) + ((lane >> 4) << 3);
                bfrag.u[v] = *(const unsigned*)&ht[wave][col][k];
            }
            #pragma unroll
            for (int mt = 0; mt < 2; ++mt) {
                FragC c{};
                c.v = __builtin_amdgcn_wmma_f32_16x16x32_bf16(
                        false, afrag[mt].v, false, bfrag.v, (short)0, c.v, false, false);
                #pragma unroll
                for (int v = 0; v < 8; ++v) {
                    int row = mt * 16 + v + ((lane >> 4) << 3);
                    float val = c.f[v];
                    float e = val > 0.f ? val : (__expf(val) - 1.f);
                    xcat[wave][row][head * HD + col] = tobf(e);
                }
            }
        }
        __syncthreads();   // ht/attnb reused next head
    }

    // ---------------- big GEMM: xcat(32x256) x W_out(256x64) ----------------
    FragC acc[2][4];
    #pragma unroll
    for (int mt = 0; mt < 2; ++mt)
        #pragma unroll
        for (int nt = 0; nt < 4; ++nt)
            acc[mt][nt] = FragC{};

    for (int kt = 0; kt < 8; ++kt) {
        FragAB a2[2];
        #pragma unroll
        for (int mt = 0; mt < 2; ++mt) {
            int row = mt * 16 + (lane & 15);
            #pragma unroll
            for (int v = 0; v < 8; ++v) {
                int k = kt * 32 + 2 * v + (v >= 4 ? 8 : 0) + ((lane >> 4) << 3);
                a2[mt].u[v] = *(const unsigned*)&xcat[wave][row][k];
            }
        }
        #pragma unroll
        for (int nt = 0; nt < 4; ++nt) {
            FragAB b2;
            int col = nt * 16 + (lane & 15);
            #pragma unroll
            for (int v = 0; v < 8; ++v) {
                int k = kt * 32 + 2 * v + (v >= 4 ? 8 : 0) + ((lane >> 4) << 3);
                b2.u[v] = *(const unsigned*)&Wt[col * 256 + k];
            }
            #pragma unroll
            for (int mt = 0; mt < 2; ++mt)
                acc[mt][nt].v = __builtin_amdgcn_wmma_f32_16x16x32_bf16(
                        false, a2[mt].v, false, b2.v, (short)0, acc[mt][nt].v, false, false);
        }
    }
    __syncthreads();   // about to overwrite ht

    // h2 -> ht (bf16, d-major) for second attention
    #pragma unroll
    for (int mt = 0; mt < 2; ++mt)
        #pragma unroll
        for (int nt = 0; nt < 4; ++nt)
            #pragma unroll
            for (int v = 0; v < 8; ++v) {
                int row = mt * 16 + v + ((lane >> 4) << 3);
                int col = nt * 16 + (lane & 15);
                ht[wave][col][row] = tobf(acc[mt][nt].f[v]);
            }
    __syncthreads();

    // ---------------- second attention (a_out) ----------------
    if (lane < NJ) {
        float s1 = 0.f, s2 = 0.f;
        for (int d = 0; d < HD; ++d) {
            float hv = (float)ht[wave][d][lane];
            s1 += hv * aout[d];
            s2 += hv * aout[HD + d];
        }
        f1a[wave][lane] = s1;
        f2a[wave][lane] = s2;
    }
    __syncthreads();

    if (lane < NJ) {
        float fi = f1a[wave][lane];
        float m = -1e30f;
        for (int j = 0; j < NJ; ++j) {
            float v = fi + f2a[wave][j];
            float lr = v > 0.f ? v : LALPHA * v;
            m = fmaxf(m, lr);
        }
        float s = 0.f;
        for (int j = 0; j < NJ; ++j) {
            float v = fi + f2a[wave][j];
            float lr = v > 0.f ? v : LALPHA * v;
            s += __expf(lr - m);
        }
        float inv = 1.f / s;
        for (int j = 0; j < NJ; ++j) {
            float v = fi + f2a[wave][j];
            float lr = v > 0.f ? v : LALPHA * v;
            attnb[wave][lane][j] = tobf(__expf(lr - m) * inv);
        }
    }
    __syncthreads();

    {
        FragAB afrag[2];
        #pragma unroll
        for (int mt = 0; mt < 2; ++mt) {
            int row = mt * 16 + (lane & 15);
            #pragma unroll
            for (int v = 0; v < 8; ++v) {
                int k = 2 * v + (v >= 4 ? 8 : 0) + ((lane >> 4) << 3);
                afrag[mt].u[v] = *(const unsigned*)&attnb[wave][row][k];
            }
        }
        #pragma unroll
        for (int nt = 0; nt < 4; ++nt) {
            FragAB bfrag;
            int col = nt * 16 + (lane & 15);
            #pragma unroll
            for (int v = 0; v < 8; ++v) {
                int k = 2 * v + (v >= 4 ? 8 : 0) + ((lane >> 4) << 3);
                bfrag.u[v] = *(const unsigned*)&ht[wave][col][k];
            }
            #pragma unroll
            for (int mt = 0; mt < 2; ++mt) {
                FragC c{};
                c.v = __builtin_amdgcn_wmma_f32_16x16x32_bf16(
                        false, afrag[mt].v, false, bfrag.v, (short)0, c.v, false, false);
                #pragma unroll
                for (int v = 0; v < 8; ++v) {
                    int row = mt * 16 + v + ((lane >> 4) << 3);
                    float val = c.f[v];
                    obuf[wave][row][nt * 16 + (lane & 15)] =
                        val > 0.f ? val : (__expf(val) - 1.f);
                }
            }
        }
    }
    __syncthreads();

    // log_softmax over d (per row j)
    if (lane < NJ) {
        float m = -1e30f;
        for (int d = 0; d < HD; ++d) m = fmaxf(m, obuf[wave][lane][d]);
        float s = 0.f;
        for (int d = 0; d < HD; ++d) s += __expf(obuf[wave][lane][d] - m);
        rowl[wave][lane] = m + __logf(s);
    }
    __syncthreads();

    // write out
    float* op = out + (size_t)t * OUTC;
    for (int idx = lane; idx < OUTC; idx += 32) {
        int j = idx >> 6, d = idx & 63;
        op[idx] = obuf[wave][j][d] - rowl[wave][j];
    }
}

extern "C" void kernel_launch(void* const* d_in, const int* in_sizes, int n_in,
                              void* d_out, int out_size, void* d_ws, size_t ws_size,
                              hipStream_t stream) {
    (void)in_sizes; (void)n_in; (void)out_size; (void)ws_size;
    const float* inp  = (const float*)d_in[0];
    // d_in[1] = seq_start_end (unused by the reference math)
    const float* Wh   = (const float*)d_in[2];
    const float* ah   = (const float*)d_in[3];
    const float* Wout = (const float*)d_in[4];
    const float* aout = (const float*)d_in[5];
    __bf16* Wt = (__bf16*)d_ws;                 // 64*256*2 = 32KB scratch
    float* out = (float*)d_out;

    prep_wout<<<64, 256, 0, stream>>>(Wout, Wt);
    gat_kernel<<<NTOK / 2, 64, 0, stream>>>(inp, Wh, ah, aout, Wt, out);
}
